// Net_10428180595026
// MI455X (gfx1250) — matrix operand order repaired and checked by reading, not compiled, not run on hardware
//
#include <hip/hip_runtime.h>
#include <hip/hip_bf16.h>

typedef __attribute__((ext_vector_type(16))) __bf16 v16bf;
typedef __attribute__((ext_vector_type(8)))  __bf16 v8bf;
typedef __attribute__((ext_vector_type(4)))  __bf16 v4bf;
typedef __attribute__((ext_vector_type(8)))  float  v8f;

#define EMB 128
#define NPG 128
#define ITERS 8

__device__ __forceinline__ float lrelu_f(float v){ return v > 0.f ? v : 0.01f*v; }

__device__ __forceinline__ unsigned hashu(unsigned a, unsigned b){
    unsigned h = a*0x9E3779B9u ^ (b + 0x85EBCA6Bu + (a<<6) + (a>>2));
    h ^= h >> 16; h *= 0x7FEB352Du;
    h ^= h >> 15; h *= 0x846CA68Bu;
    h ^= h >> 16;
    return h;
}

// ---------------------------------------------------------------------------
// WMMA fragment loaders (layouts per CDNA5 ISA 7.12.2)
// A 16x32 bf16: lane<16 -> M=lane,   K in {0..7, 16..23} (+32*kt)
//               lane>=16 -> M=lane-16, K in {8..15, 24..31}
// B 32x16 bf16 (packed): fragment (kt,nt) = 32 lanes x 16 contiguous bf16
// ---------------------------------------------------------------------------
__device__ __forceinline__ v16bf load_afrag(const __bf16* __restrict__ X,
                                            int row, int kt, int kb){
    const v8bf* ap = (const v8bf*)(X + (size_t)row*EMB + kt*32 + kb);
    v8bf lo = ap[0];               // K = 32kt+kb+0..7
    v8bf hi = ap[2];               // K = 32kt+kb+16..23
    v16bf a;
    #pragma unroll
    for (int i = 0; i < 8; ++i){ a[i] = lo[i]; a[8+i] = hi[i]; }
    return a;
}

// same fragment, but source matrix is f32 (converted to bf16 in-register)
__device__ __forceinline__ v16bf load_afrag_f32(const float* __restrict__ X,
                                                int row, int kt, int kb){
    const v8f* ap = (const v8f*)(X + (size_t)row*EMB + kt*32 + kb);
    v8f lo = ap[0];
    v8f hi = ap[2];
    v16bf a;
    #pragma unroll
    for (int i = 0; i < 8; ++i){ a[i] = (__bf16)lo[i]; a[8+i] = (__bf16)hi[i]; }
    return a;
}

__device__ __forceinline__ v16bf load_bfrag(const __bf16* __restrict__ Wp,
                                            int kt, int nt, int lane){
    return *(const v16bf*)(Wp + ((((size_t)kt*8 + nt)*32 + lane) << 4));
}

// Pack a 128x128 row-major (K x N) f32 weight into bf16 B-fragment layout:
// lanes 0-15 col N=lane hold K=base+{0..7,16..23}, lanes 16-31 hold K=base+{8..15,24..31}.
__global__ void pack_w_kernel(const float* __restrict__ W, __bf16* __restrict__ Wp){
    int t = blockIdx.x*256 + threadIdx.x;          // 0..16383
    int e    =  t        & 15;
    int lane = (t >> 4)  & 31;
    int nt   = (t >> 9)  & 7;
    int kt   = (t >> 12) & 3;
    int k = kt*32 + ((e >> 3) << 4) + ((lane >> 4) << 3) + (e & 7);
    int n = nt*16 + (lane & 15);
    Wp[t] = (__bf16)W[k*EMB + n];
}

__global__ void zero_kernel(float* __restrict__ p, long long n){
    long long t = (long long)blockIdx.x*256 + threadIdx.x;
    if (t < n) p[t] = 0.f;
}

// x = lrelu(node_feats @ W_embed + b) ; K=5 so scalar FMAs, one thread/elem
__global__ void embed_kernel(const float* __restrict__ nf, const float* __restrict__ W,
                             const float* __restrict__ b, float* __restrict__ x,
                             __bf16* __restrict__ xbf, int N){
    long long t = (long long)blockIdx.x*256 + threadIdx.x;
    if (t >= (long long)N*EMB) return;
    int c = (int)(t & 127); long long n = t >> 7;
    float s = b[c];
    #pragma unroll
    for (int f = 0; f < 5; ++f) s += nf[n*5 + f]*W[f*EMB + c];
    s = lrelu_f(s);
    x[t] = s; xbf[t] = (__bf16)s;
}

// ---------------------------------------------------------------------------
// Fused GEMM: C = act( X1@W1 [+ X2f@W2] [+ addGrp[row>>7]] [+ bias] )
// Compile-time specialized: no branches in the K-loop or epilogue.
// Block = 256 threads = 8 waves; wave -> 16-row x 64-col tile; grid = rows/64.
// X2 stream is f32-sourced (converted in-register) to avoid a cvt pass.
// ---------------------------------------------------------------------------
template<bool DUAL, bool GRP, bool BIAS, bool ACT, bool CBF>
__global__ void __launch_bounds__(256) gemm128_kernel(
    const __bf16* __restrict__ X1, const __bf16* __restrict__ W1p,
    const float*  __restrict__ X2f, const __bf16* __restrict__ W2p,
    const float* __restrict__ addGrp, const float* __restrict__ bias,
    float* __restrict__ Cf, __bf16* __restrict__ Cbf)
{
    int lane = threadIdx.x & 31;
    int wave = threadIdx.x >> 5;      // 0..7
    int rt   = wave >> 1;             // row tile in block
    int ch   = wave & 1;              // column half
    int tileRow = blockIdx.x*64 + rt*16;
    int arow = tileRow + (lane & 15);
    int kb   = (lane >> 4) << 3;

    v8f acc[4];
    #pragma unroll
    for (int j = 0; j < 4; ++j) acc[j] = {0.f,0.f,0.f,0.f,0.f,0.f,0.f,0.f};

    __builtin_prefetch(X1 + (size_t)arow*EMB, 0, 0);

    #pragma unroll
    for (int kt = 0; kt < 4; ++kt){
        v16bf a1 = load_afrag(X1, arow, kt, kb);
        v16bf a2;
        if constexpr (DUAL) a2 = load_afrag_f32(X2f, arow, kt, kb);
        #pragma unroll
        for (int j = 0; j < 4; ++j){
            int nt = ch*4 + j;
            v16bf b1 = load_bfrag(W1p, kt, nt, lane);
            acc[j] = __builtin_amdgcn_wmma_f32_16x16x32_bf16(
                false, a1, false, b1, (short)0, acc[j], false, false);
            if constexpr (DUAL){
                v16bf b2 = load_bfrag(W2p, kt, nt, lane);
                acc[j] = __builtin_amdgcn_wmma_f32_16x16x32_bf16(
                    false, a2, false, b2, (short)0, acc[j], false, false);
            }
        }
    }

    // C/D layout: lane -> N = lane&15, VGPR r -> M = r + (lane>>4)*8
    int mbase = (lane >> 4) << 3;
    int ncol  = lane & 15;
    #pragma unroll
    for (int j = 0; j < 4; ++j){
        int col = (ch*4 + j)*16 + ncol;
        float bv = 0.f;
        if constexpr (BIAS) bv = bias[col];
        #pragma unroll
        for (int r = 0; r < 8; ++r){
            int row = tileRow + mbase + r;
            size_t idx = (size_t)row*EMB + col;
            float v = acc[j][r];
            if constexpr (GRP) v += addGrp[(size_t)(row >> 7)*EMB + col];
            if constexpr (BIAS) v += bv;
            if constexpr (ACT)  v = lrelu_f(v);
            Cf[idx] = v;
            if constexpr (CBF) Cbf[idx] = (__bf16)v;
        }
    }
}

// ---------------------------------------------------------------------------
// Dual-output message GEMM: A = X@Wsrc + b_msg, B = X@Wdst (both bf16 out).
// Shares one A-fragment stream across both weight matrices (8 WMMA / k-step).
// ---------------------------------------------------------------------------
__global__ void __launch_bounds__(256) msg_gemm_kernel(
    const __bf16* __restrict__ Xbf, const __bf16* __restrict__ Wsp,
    const __bf16* __restrict__ Wdp, const float* __restrict__ bmsg,
    __bf16* __restrict__ Abf, __bf16* __restrict__ Bbf)
{
    int lane = threadIdx.x & 31;
    int wave = threadIdx.x >> 5;
    int rt   = wave >> 1;
    int ch   = wave & 1;
    int tileRow = blockIdx.x*64 + rt*16;
    int arow = tileRow + (lane & 15);
    int kb   = (lane >> 4) << 3;

    v8f accA[4], accB[4];
    #pragma unroll
    for (int j = 0; j < 4; ++j){
        accA[j] = {0.f,0.f,0.f,0.f,0.f,0.f,0.f,0.f};
        accB[j] = {0.f,0.f,0.f,0.f,0.f,0.f,0.f,0.f};
    }

    #pragma unroll
    for (int kt = 0; kt < 4; ++kt){
        v16bf a = load_afrag(Xbf, arow, kt, kb);
        #pragma unroll
        for (int j = 0; j < 4; ++j){
            int nt = ch*4 + j;
            v16bf bs = load_bfrag(Wsp, kt, nt, lane);
            accA[j] = __builtin_amdgcn_wmma_f32_16x16x32_bf16(
                false, a, false, bs, (short)0, accA[j], false, false);
            v16bf bd = load_bfrag(Wdp, kt, nt, lane);
            accB[j] = __builtin_amdgcn_wmma_f32_16x16x32_bf16(
                false, a, false, bd, (short)0, accB[j], false, false);
        }
    }

    int mbase = (lane >> 4) << 3;
    int ncol  = lane & 15;
    #pragma unroll
    for (int j = 0; j < 4; ++j){
        int col = (ch*4 + j)*16 + ncol;
        float bm = bmsg[col];
        #pragma unroll
        for (int r = 0; r < 8; ++r){
            int row = tileRow + mbase + r;
            size_t idx = (size_t)row*EMB + col;
            Abf[idx] = (__bf16)(accA[j][r] + bm);
            Bbf[idx] = (__bf16)accB[j][r];
        }
    }
}

// per-edge: m = lrelu(A[src]+B[dst]+ea0*We0+ea1*We1), agg[dst] += m (atomics)
// 32 threads per edge, 4 cols per thread; A/B gathered in bf16 (half bytes)
__global__ void edge_kernel(const int* __restrict__ ei, const float* __restrict__ ea,
                            const __bf16* __restrict__ A, const __bf16* __restrict__ B,
                            const float* __restrict__ We, float* __restrict__ agg,
                            long long E){
    long long t = (long long)blockIdx.x*256 + threadIdx.x;
    long long e = t >> 5;
    if (e >= E) return;
    int c0 = ((int)t & 31) << 2;
    int src = ei[e], dst = ei[E + e];
    float ea0 = ea[e*2], ea1 = ea[e*2 + 1];
    v4bf a4 = *(const v4bf*)(A + (size_t)src*EMB + c0);
    v4bf b4 = *(const v4bf*)(B + (size_t)dst*EMB + c0);
    float4 w0 = *(const float4*)(We + c0);
    float4 w1 = *(const float4*)(We + EMB + c0);
    float* d = agg + (size_t)dst*EMB + c0;
    atomicAdd(d + 0, lrelu_f((float)a4[0] + (float)b4[0] + ea0*w0.x + ea1*w1.x));
    atomicAdd(d + 1, lrelu_f((float)a4[1] + (float)b4[1] + ea0*w0.y + ea1*w1.y));
    atomicAdd(d + 2, lrelu_f((float)a4[2] + (float)b4[2] + ea0*w0.z + ea1*w1.z));
    atomicAdd(d + 3, lrelu_f((float)a4[3] + (float)b4[3] + ea0*w0.w + ea1*w1.w));
}

// pooled[g,c] = mean_i x[g*128+i, c]   (coalesced over c)
__global__ void pool_kernel(const float* __restrict__ x, float* __restrict__ out,
                            __bf16* __restrict__ outbf, int G){
    int t = blockIdx.x*256 + threadIdx.x;
    if (t >= G*EMB) return;
    int c = t & 127, g = t >> 7;
    const float* p = x + (size_t)g*NPG*EMB + c;
    float s = 0.f;
    #pragma unroll 4
    for (int i = 0; i < NPG; ++i) s += p[(size_t)i*EMB];
    s *= (1.f/NPG);
    out[t] = s;
    if (outbf) outbf[t] = (__bf16)s;
}

// per-graph: action logits, softmax, categorical sample, value head
__global__ void action_kernel(const float* __restrict__ g, const float* __restrict__ Was,
                              const float* __restrict__ bas, const float* __restrict__ Wv,
                              const float* __restrict__ bv, float* __restrict__ out,
                              float* __restrict__ act_sel, float* __restrict__ a_prob, int G){
    __shared__ float sl[5][128];
    int gid = blockIdx.x, c = threadIdx.x;
    float gv = g[(size_t)gid*EMB + c];
    #pragma unroll
    for (int a = 0; a < 4; ++a) sl[a][c] = gv * Was[c*4 + a];
    sl[4][c] = gv * Wv[c];
    __syncthreads();
    for (int s = 64; s > 0; s >>= 1){
        if (c < s){
            #pragma unroll
            for (int a = 0; a < 5; ++a) sl[a][c] += sl[a][c + s];
        }
        __syncthreads();
    }
    if (c == 0){
        float lg[4], p[4], mx = -1e30f;
        #pragma unroll
        for (int a = 0; a < 4; ++a){ lg[a] = sl[a][0] + bas[a]; mx = fmaxf(mx, lg[a]); }
        float se = 0.f;
        #pragma unroll
        for (int a = 0; a < 4; ++a){ p[a] = __expf(lg[a] - mx); se += p[a]; }
        #pragma unroll
        for (int a = 0; a < 4; ++a) p[a] /= se;
        float u = (hashu(42u, (unsigned)gid) & 0xFFFFFFu) * (1.f/16777216.f);
        int sel = 3; float cum = 0.f;
        #pragma unroll
        for (int a = 0; a < 4; ++a){ cum += p[a]; if (u < cum){ sel = a; break; } }
        out[gid]        = (float)sel;          // action_selected
        out[2*G + gid]  = sl[4][0] + bv[0];    // value
        act_sel[gid] = (float)sel;
        a_prob[gid]  = p[sel];
    }
}

// per-graph: per-node logit for selected action, scatter-softmax, Gumbel-max
__global__ void node_kernel(const float* __restrict__ x, const float* __restrict__ Wns,
                            const float* __restrict__ bns, const float* __restrict__ act_sel,
                            const float* __restrict__ a_prob, float* __restrict__ out, int G){
    __shared__ float sv[128];
    __shared__ float sp[128];
    __shared__ int   si[128];
    int gid = blockIdx.x, i = threadIdx.x;
    long long n = (long long)gid*NPG + i;
    int a = (int)act_sel[gid];
    const float* xr = x + (size_t)n*EMB;
    float lg = bns[a];
    #pragma unroll 4
    for (int c = 0; c < EMB; ++c) lg += xr[c]*Wns[c*4 + a];
    sv[i] = lg; __syncthreads();
    for (int s = 64; s > 0; s >>= 1){ if (i < s) sv[i] = fmaxf(sv[i], sv[i + s]); __syncthreads(); }
    float mx = sv[0]; __syncthreads();
    float e = __expf(lg - mx);
    sv[i] = e; __syncthreads();
    for (int s = 64; s > 0; s >>= 1){ if (i < s) sv[i] += sv[i + s]; __syncthreads(); }
    float denom = sv[0]; __syncthreads();
    float sm = e / denom;
    sp[i] = sm;
    float u = (hashu(0x002A2A2Au, (unsigned)n) & 0xFFFFFFu) * (1.f/16777216.f) + 1e-9f;
    float gum = -__logf(-__logf(u));
    sv[i] = __logf(sm + 1e-20f) + gum; si[i] = i;
    __syncthreads();
    for (int s = 64; s > 0; s >>= 1){
        if (i < s && sv[i + s] > sv[i]){ sv[i] = sv[i + s]; si[i] = si[i + s]; }
        __syncthreads();
    }
    if (i == 0){
        int sel = si[0];
        out[G + gid]   = (float)sel;                 // node_selected
        out[3*G + gid] = a_prob[gid] * sp[sel];      // tot_prob
    }
}

extern "C" void kernel_launch(void* const* d_in, const int* in_sizes, int n_in,
                              void* d_out, int out_size, void* d_ws, size_t ws_size,
                              hipStream_t stream){
    const float* nf        = (const float*)d_in[0];
    const float* ea        = (const float*)d_in[1];
    const int*   ei        = (const int*)  d_in[2];
    const float* W_embed   = (const float*)d_in[3];
    const float* b_embed   = (const float*)d_in[4];
    const float* W_msg_src = (const float*)d_in[5];
    const float* W_msg_dst = (const float*)d_in[6];
    const float* W_msg_e   = (const float*)d_in[7];
    const float* b_msg     = (const float*)d_in[8];
    const float* W_upd_x   = (const float*)d_in[9];
    const float* W_upd_m   = (const float*)d_in[10];
    const float* W_upd_g   = (const float*)d_in[11];
    const float* b_upd     = (const float*)d_in[12];
    const float* W_glob_g  = (const float*)d_in[13];
    const float* W_glob_p  = (const float*)d_in[14];
    const float* b_glob    = (const float*)d_in[15];
    const float* W_ns      = (const float*)d_in[16];
    const float* b_ns      = (const float*)d_in[17];
    const float* W_as      = (const float*)d_in[18];
    const float* b_as      = (const float*)d_in[19];
    const float* W_v       = (const float*)d_in[20];
    const float* b_v       = (const float*)d_in[21];

    const int       N = in_sizes[0] / 5;
    const long long E = in_sizes[1] / 2;
    const int       G = N / NPG;
    const long long NE = (long long)N * EMB;
    const long long GE = (long long)G * EMB;

    char* wp = (char*)d_ws;
    auto carve = [&](size_t bytes)->void*{
        void* r = (void*)wp; wp += (bytes + 255) & ~(size_t)255; return r;
    };
    float*  x       = (float*) carve(NE*4);
    __bf16* xbf     = (__bf16*)carve(NE*2);
    __bf16* Abf     = (__bf16*)carve(NE*2);
    __bf16* Bbf     = (__bf16*)carve(NE*2);
    float*  agg     = (float*) carve(NE*4);
    float*  g       = (float*) carve(GE*4);
    __bf16* gbf     = (__bf16*)carve(GE*2);
    float*  pooled  = (float*) carve(GE*4);
    float*  Gu      = (float*) carve(GE*4);
    float*  act_sel = (float*) carve((size_t)G*4);
    float*  a_prob  = (float*) carve((size_t)G*4);
    __bf16* Wp_src  = (__bf16*)carve(16384*2);
    __bf16* Wp_dst  = (__bf16*)carve(16384*2);
    __bf16* Wp_updx = (__bf16*)carve(16384*2);
    __bf16* Wp_updm = (__bf16*)carve(16384*2);
    __bf16* Wp_updg = (__bf16*)carve(16384*2);
    __bf16* Wp_glg  = (__bf16*)carve(16384*2);
    __bf16* Wp_glp  = (__bf16*)carve(16384*2);

    // pack all 128x128 weights into WMMA B-fragment bf16 layout
    pack_w_kernel<<<64, 256, 0, stream>>>(W_msg_src, Wp_src);
    pack_w_kernel<<<64, 256, 0, stream>>>(W_msg_dst, Wp_dst);
    pack_w_kernel<<<64, 256, 0, stream>>>(W_upd_x,  Wp_updx);
    pack_w_kernel<<<64, 256, 0, stream>>>(W_upd_m,  Wp_updm);
    pack_w_kernel<<<64, 256, 0, stream>>>(W_upd_g,  Wp_updg);
    pack_w_kernel<<<64, 256, 0, stream>>>(W_glob_g, Wp_glg);
    pack_w_kernel<<<64, 256, 0, stream>>>(W_glob_p, Wp_glp);

    // embed + initial global = mean pool
    embed_kernel<<<(int)((NE + 255)/256), 256, 0, stream>>>(nf, W_embed, b_embed, x, xbf, N);
    pool_kernel<<<(int)((GE + 255)/256), 256, 0, stream>>>(x, g, gbf, G);

    const int gridN = N/64, gridG = G/64;

    for (int it = 0; it < ITERS; ++it){
        // hoisted message GEMMs: A = x@Ws + b_msg, B = x@Wd (bf16 outputs)
        msg_gemm_kernel<<<gridN, 256, 0, stream>>>(xbf, Wp_src, Wp_dst, b_msg, Abf, Bbf);
        // edge scatter: agg = segsum(lrelu(A[src]+B[dst]+e@We))
        zero_kernel<<<(int)((NE + 255)/256), 256, 0, stream>>>(agg, NE);
        edge_kernel<<<(int)((E*32 + 255)/256), 256, 0, stream>>>(ei, ea, Abf, Bbf, W_msg_e, agg, E);
        // Gu = g@W_upd_g  (per-graph broadcast term of the node update)
        gemm128_kernel<false,false,false,false,false><<<gridG, 256, 0, stream>>>(
            gbf, Wp_updg, nullptr, nullptr, nullptr, nullptr, Gu, nullptr);
        // node update: x = lrelu(x@Wx + agg@Wm + Gu[batch] + b)  (fused 2-input GEMM)
        gemm128_kernel<true,true,true,true,true><<<gridN, 256, 0, stream>>>(
            xbf, Wp_updx, agg, Wp_updm, Gu, b_upd, x, xbf);
        // global update: g = lrelu(g@Wg + pooled@Wp + b)  (fused 2-input GEMM)
        pool_kernel<<<(int)((GE + 255)/256), 256, 0, stream>>>(x, pooled, nullptr, G);
        gemm128_kernel<true,false,true,true,true><<<gridG, 256, 0, stream>>>(
            gbf, Wp_glg, pooled, Wp_glp, nullptr, b_glob, g, gbf);
    }

    // heads + sampling; d_out = [action(G) | node(G) | value(G) | tot_prob(G)]
    action_kernel<<<G, 128, 0, stream>>>(g, W_as, b_as, W_v, b_v, (float*)d_out, act_sel, a_prob, G);
    node_kernel  <<<G, 128, 0, stream>>>(x, W_ns, b_ns, act_sel, a_prob, (float*)d_out, G);
}
